// FeatureAttention_90323162235563
// MI455X (gfx1250) — matrix operand order repaired and checked by reading, not compiled, not run on hardware
//
#include <hip/hip_runtime.h>

// ---------------------------------------------------------------------------
// FeatureAttention for MI455X (gfx1250, wave32, WMMA + async-to-LDS)
//   B=4, T=2048, D=2048, H=16, HO=128
// Compute-bound: 4 big GEMMs (275 GFLOP) -> v_wmma_f32_16x16x32_bf16 with
// bf16 hi/lo split (3 WMMAs per k-step). Tiles staged with
// GLOBAL_LOAD_ASYNC_TO_LDS_B128 into double-buffered LDS (ASYNCcnt pipelined).
// Block tile 128x128, wave tile 32x64 -> 24 WMMAs : 24 ds_loads per k-step.
// ---------------------------------------------------------------------------

typedef __bf16 v16bf __attribute__((ext_vector_type(16)));
typedef float  v8f   __attribute__((ext_vector_type(8)));

union Frag16 {               // 16 bf16 elements = 8 VGPRs
    v16bf        v;
    unsigned int u[8];
};

static __device__ __forceinline__ unsigned short f32_to_bf16_rne(float a) {
    unsigned int u = __float_as_uint(a);
    u += 0x7FFFu + ((u >> 16) & 1u);   // round to nearest even
    return (unsigned short)(u >> 16);
}

// 16B/lane async copy global -> LDS (ASYNCcnt). lds_off = low 32 bits of the
// generic LDS address (LDS aperture keeps the byte offset in addr[31:0]).
static __device__ __forceinline__ void async_load_b128(unsigned lds_off, const void* gptr) {
    asm volatile("global_load_async_to_lds_b128 %0, %1, off"
                 :: "v"(lds_off), "v"(gptr) : "memory");
}
static __device__ __forceinline__ void wait_async_le8() {
    asm volatile("s_wait_asynccnt 0x8" ::: "memory");
}
static __device__ __forceinline__ void wait_async_0() {
    asm volatile("s_wait_asynccnt 0x0" ::: "memory");
}
static __device__ __forceinline__ unsigned lds_off32(const void* p) {
    return (unsigned)(unsigned long long)p;
}

// ---------------------------------------------------------------------------
// fp32 -> (hi, lo) bf16 split:  a ~= hi + lo  (16+ effective mantissa bits)
// ---------------------------------------------------------------------------
__global__ __launch_bounds__(256) void split_bf16(
    const float* __restrict__ src,
    unsigned short* __restrict__ hi,
    unsigned short* __restrict__ lo,
    long long n)
{
    long long i = (long long)blockIdx.x * blockDim.x + threadIdx.x;
    if (i >= n) return;
    float a  = src[i];
    unsigned short h = f32_to_bf16_rne(a);
    float hf = __uint_as_float((unsigned int)h << 16);
    hi[i] = h;
    lo[i] = f32_to_bf16_rne(a - hf);
}

// ---------------------------------------------------------------------------
// fp32 W[K][N] -> transposed bf16 hi/lo WT[N][K] (LDS 32x32 tiles, coalesced
// both directions). Done once per weight; lets the GEMM stage B tiles as
// contiguous rows (no LDS scatter stores).
// ---------------------------------------------------------------------------
__global__ __launch_bounds__(256) void split_transpose_bf16(
    const float* __restrict__ src,          // [Kd][Nd]
    unsigned short* __restrict__ hiT,       // [Nd][Kd]
    unsigned short* __restrict__ loT,       // [Nd][Kd]
    int Kd, int Nd)
{
    __shared__ float t[32][33];
    const int n0 = blockIdx.x * 32;
    const int k0 = blockIdx.y * 32;
    for (int i = threadIdx.x; i < 1024; i += 256) {
        const int kk = i >> 5, nn = i & 31;
        t[kk][nn] = src[(size_t)(k0 + kk) * Nd + n0 + nn];
    }
    __syncthreads();
    for (int i = threadIdx.x; i < 1024; i += 256) {
        const int nn = i >> 5, kk = i & 31;
        const float a = t[kk][nn];
        const unsigned short h = f32_to_bf16_rne(a);
        const float hf = __uint_as_float((unsigned int)h << 16);
        hiT[(size_t)(n0 + nn) * Kd + k0 + kk] = h;
        loT[(size_t)(n0 + nn) * Kd + k0 + kk] = f32_to_bf16_rne(a - hf);
    }
}

// ---------------------------------------------------------------------------
// C[M,N] (fp32) = A[M,K] x B[K,N]; A row-major bf16 hi/lo, B given TRANSPOSED
// (BT[N][K]) bf16 hi/lo. Block tile 128x128, BK=32, 256 threads = 8 waves
// (4 in M x 2 in N), each wave computes 32x64 = 2x4 WMMA accumulators.
// Double-buffered LDS filled by global_load_async_to_lds_b128 (8 issues per
// wave per stage), pipelined with s_wait_asynccnt 0x8.
// ---------------------------------------------------------------------------
__global__ __launch_bounds__(256) void gemm_bf16x2(
    const unsigned short* __restrict__ Ahg,  // [M,K] bf16-hi
    const unsigned short* __restrict__ Alg,  // [M,K] bf16-lo
    const unsigned short* __restrict__ Bht,  // [N,K] bf16-hi (transposed)
    const unsigned short* __restrict__ Blt,  // [N,K] bf16-lo (transposed)
    float* __restrict__ C,                   // [M,N]
    const float* __restrict__ bias,          // [N] or nullptr
    int M, int N, int K)
{
    __shared__ alignas(16) unsigned short Ash[2][128][32];
    __shared__ alignas(16) unsigned short Asl[2][128][32];
    __shared__ alignas(16) unsigned short Bsh[2][128][32];   // [n][k]
    __shared__ alignas(16) unsigned short Bsl[2][128][32];

    const int tid   = threadIdx.x;
    const int wid   = tid >> 5;
    const int lane  = tid & 31;
    const int waveM = wid & 3;          // 0..3  -> 32-row slab
    const int waveN = wid >> 2;         // 0..1  -> 64-col slab
    const int lm    = lane & 15;        // m (or n) within 16x16 tile
    const int lh    = lane >> 4;        // half selects K-range / row group

    const int m0 = blockIdx.y * 128;
    const int n0 = blockIdx.x * 128;

    // 16B-chunk mapping for 128x32-ushort tiles: row = c>>2, seg = (c&3)*8
    const int r0 = tid >> 2,          s0 = (tid & 3) << 3;        // chunks 0..255
    const int r1 = (tid + 256) >> 2;                              // chunks 256..511

    // per wave per stage: exactly 8 async b128 instructions
    auto issue_tile = [&](int b, int kk0) {
        async_load_b128(lds_off32(&Ash[b][r0][s0]), Ahg + (size_t)(m0 + r0) * K + kk0 + s0);
        async_load_b128(lds_off32(&Ash[b][r1][s0]), Ahg + (size_t)(m0 + r1) * K + kk0 + s0);
        async_load_b128(lds_off32(&Asl[b][r0][s0]), Alg + (size_t)(m0 + r0) * K + kk0 + s0);
        async_load_b128(lds_off32(&Asl[b][r1][s0]), Alg + (size_t)(m0 + r1) * K + kk0 + s0);
        async_load_b128(lds_off32(&Bsh[b][r0][s0]), Bht + (size_t)(n0 + r0) * K + kk0 + s0);
        async_load_b128(lds_off32(&Bsh[b][r1][s0]), Bht + (size_t)(n0 + r1) * K + kk0 + s0);
        async_load_b128(lds_off32(&Bsl[b][r0][s0]), Blt + (size_t)(n0 + r0) * K + kk0 + s0);
        async_load_b128(lds_off32(&Bsl[b][r1][s0]), Blt + (size_t)(n0 + r1) * K + kk0 + s0);
    };

    v8f acc[2][4] = {};

    issue_tile(0, 0);
    int buf = 0;
    for (int k0 = 0; k0 < K; k0 += 32) {
        const bool more = (k0 + 32) < K;
        if (more) issue_tile(buf ^ 1, k0 + 32);
        if (more) wait_async_le8(); else wait_async_0();
        __syncthreads();

        // ---- fragment loads (ISA 16-bit A 16x32 / B 32x16 layouts) ----
        // A lane(lm,lh): e0..7 = A[m][8*lh .. +7], e8..15 = A[m][16+8*lh .. +7]
        Frag16 aH[2], aL[2], bH[4], bL[4];
        #pragma unroll
        for (int sm = 0; sm < 2; ++sm) {
            const int m  = waveM * 32 + sm * 16 + lm;
            const int kb = lh * 8;
            uint4 t0 = *(const uint4*)&Ash[buf][m][kb];
            uint4 t1 = *(const uint4*)&Ash[buf][m][16 + kb];
            aH[sm].u[0]=t0.x; aH[sm].u[1]=t0.y; aH[sm].u[2]=t0.z; aH[sm].u[3]=t0.w;
            aH[sm].u[4]=t1.x; aH[sm].u[5]=t1.y; aH[sm].u[6]=t1.z; aH[sm].u[7]=t1.w;
            t0 = *(const uint4*)&Asl[buf][m][kb];
            t1 = *(const uint4*)&Asl[buf][m][16 + kb];
            aL[sm].u[0]=t0.x; aL[sm].u[1]=t0.y; aL[sm].u[2]=t0.z; aL[sm].u[3]=t0.w;
            aL[sm].u[4]=t1.x; aL[sm].u[5]=t1.y; aL[sm].u[6]=t1.z; aL[sm].u[7]=t1.w;
        }
        // B lane(lm,lh): e0..7 = B[k=8*lh+e][n], e8..15 = B[k=16+8*lh+e-8][n]  -> transposed LDS row
        #pragma unroll
        for (int sn = 0; sn < 4; ++sn) {
            const int n  = waveN * 64 + sn * 16 + lm;
            const int kb = lh * 8;
            uint4 t0 = *(const uint4*)&Bsh[buf][n][kb];
            uint4 t1 = *(const uint4*)&Bsh[buf][n][16 + kb];
            bH[sn].u[0]=t0.x; bH[sn].u[1]=t0.y; bH[sn].u[2]=t0.z; bH[sn].u[3]=t0.w;
            bH[sn].u[4]=t1.x; bH[sn].u[5]=t1.y; bH[sn].u[6]=t1.z; bH[sn].u[7]=t1.w;
            t0 = *(const uint4*)&Bsl[buf][n][kb];
            t1 = *(const uint4*)&Bsl[buf][n][16 + kb];
            bL[sn].u[0]=t0.x; bL[sn].u[1]=t0.y; bL[sn].u[2]=t0.z; bL[sn].u[3]=t0.w;
            bL[sn].u[4]=t1.x; bL[sn].u[5]=t1.y; bL[sn].u[6]=t1.z; bL[sn].u[7]=t1.w;
        }

        #pragma unroll
        for (int sm = 0; sm < 2; ++sm)
            #pragma unroll
            for (int sn = 0; sn < 4; ++sn) {
                acc[sm][sn] = __builtin_amdgcn_wmma_f32_16x16x32_bf16(
                    false, aH[sm].v, false, bH[sn].v, (short)0, acc[sm][sn], false, false);
                acc[sm][sn] = __builtin_amdgcn_wmma_f32_16x16x32_bf16(
                    false, aH[sm].v, false, bL[sn].v, (short)0, acc[sm][sn], false, false);
                acc[sm][sn] = __builtin_amdgcn_wmma_f32_16x16x32_bf16(
                    false, aL[sm].v, false, bH[sn].v, (short)0, acc[sm][sn], false, false);
            }
        __syncthreads();   // protect buf^1 (refilled next iteration) from WAR
        buf ^= 1;
    }

    // ---- epilogue: C/D layout lane(lm,lh) vgpr r -> M = 8*lh + r, N = lm ----
    #pragma unroll
    for (int sm = 0; sm < 2; ++sm)
        #pragma unroll
        for (int sn = 0; sn < 4; ++sn) {
            const int gn = n0 + waveN * 64 + sn * 16 + lm;
            const float bv = bias ? bias[gn] : 0.0f;
            #pragma unroll
            for (int r = 0; r < 8; ++r) {
                const int gm = m0 + waveM * 32 + sm * 16 + lh * 8 + r;
                C[(size_t)gm * N + gn] = acc[sm][sn][r] + bv;
            }
        }
}

// ---------------------------------------------------------------------------
// scores[f,g] = sum_t q[t,f] k[t,g] / sqrt(HO), softmax over g.
// One 128-thread block per (b,h,f); thread = g. (~1.5% of total FLOPs,
// q/k head slices live in the 192MB L2.)
// ---------------------------------------------------------------------------
__global__ __launch_bounds__(128) void scores_softmax(
    const float* __restrict__ q, const float* __restrict__ k,
    float* __restrict__ w, int Bb, int T, int D, int H, int HO)
{
    const int f = blockIdx.x, h = blockIdx.y, b = blockIdx.z;
    const int g = threadIdx.x;

    const float* qp = q + (size_t)b * T * D + h * HO + f;
    const float* kp = k + (size_t)b * T * D + h * HO + g;
    float acc = 0.0f;
    for (int t = 0; t < T; ++t)
        acc += qp[(size_t)t * D] * kp[(size_t)t * D];
    acc *= rsqrtf((float)HO);

    __shared__ float red[128];
    red[g] = acc;
    __syncthreads();
    for (int s = 64; s > 0; s >>= 1) {
        if (g < s) red[g] = fmaxf(red[g], red[g + s]);
        __syncthreads();
    }
    const float mx = red[0];
    __syncthreads();
    const float e = __expf(acc - mx);
    red[g] = e;
    __syncthreads();
    for (int s = 64; s > 0; s >>= 1) {
        if (g < s) red[g] += red[g + s];
        __syncthreads();
    }
    w[(((size_t)b * H + h) * HO + f) * HO + g] = e / red[0];
}

// ---------------------------------------------------------------------------
// ctx[t,f] = sum_g w[f,g] * (v[t,g] + 1/(T-t)).  One block per (b,h,t-chunk),
// thread = f; LDS-tiled over g with per-thread 16-token accumulators.
// ---------------------------------------------------------------------------
#define CTX_TCH 16
__global__ __launch_bounds__(128) void ctx_kernel(
    const float* __restrict__ w, const float* __restrict__ v,
    float* __restrict__ ctx, int Bb, int T, int D, int H, int HO)
{
    __shared__ float wch[128][33];        // padded vs bank conflicts
    __shared__ float vch[CTX_TCH][33];

    const int f  = threadIdx.x;
    const int h  = blockIdx.y, b = blockIdx.z;
    const int t0 = blockIdx.x * CTX_TCH;
    const float* wp = w + ((size_t)b * H + h) * HO * HO;

    float acc[CTX_TCH];
    #pragma unroll
    for (int i = 0; i < CTX_TCH; ++i) acc[i] = 0.0f;

    for (int gc = 0; gc < HO; gc += 32) {
        for (int i = f; i < 128 * 32; i += 128) {
            const int rf = i >> 5, rg = i & 31;
            wch[rf][rg] = wp[(size_t)rf * HO + gc + rg];
        }
        for (int i = f; i < CTX_TCH * 32; i += 128) {
            const int tt = i >> 5, rg = i & 31;
            const int t = t0 + tt;
            vch[tt][rg] = v[((size_t)b * T + t) * D + h * HO + gc + rg]
                        + 1.0f / (float)(T - t);
        }
        __syncthreads();
        #pragma unroll
        for (int tt = 0; tt < CTX_TCH; ++tt) {
            float a = 0.0f;
            #pragma unroll
            for (int g2 = 0; g2 < 32; ++g2) a += wch[f][g2] * vch[tt][g2];
            acc[tt] += a;
        }
        __syncthreads();
    }
    #pragma unroll
    for (int tt = 0; tt < CTX_TCH; ++tt)
        ctx[((size_t)b * T + t0 + tt) * D + h * HO + f] = acc[tt];
}

// ---------------------------------------------------------------------------
// host-side orchestration
// ---------------------------------------------------------------------------
extern "C" void kernel_launch(void* const* d_in, const int* in_sizes, int n_in,
                              void* d_out, int out_size, void* d_ws, size_t ws_size,
                              hipStream_t stream)
{
    constexpr int Bb = 4, T = 2048, D = 2048, H = 16, HO = 128;
    constexpr int MR = Bb * T;                       // 8192 rows

    const float* x  = (const float*)d_in[0];
    const float* Wq = (const float*)d_in[1];
    const float* Wk = (const float*)d_in[2];
    const float* Wv = (const float*)d_in[3];
    const float* Wp = (const float*)d_in[4];
    const float* bp = (const float*)d_in[5];
    float* out = (float*)d_out;

    char* ws = (char*)d_ws;
    size_t off = 0;
    auto take = [&](size_t bytes) { char* p = ws + off; off += bytes; return p; };

    float*          qbuf = (float*)take((size_t)MR * D * 4);      // later: ctx
    float*          kbuf = (float*)take((size_t)MR * D * 4);      // later: v
    unsigned short* xh   = (unsigned short*)take((size_t)MR * D * 2); // later: ctx-hi
    unsigned short* xl   = (unsigned short*)take((size_t)MR * D * 2); // later: ctx-lo
    float*          attw = (float*)take((size_t)Bb * H * HO * HO * 4);
    unsigned short* Wh   = (unsigned short*)take((size_t)D * D * 2);  // [N][K] transposed
    unsigned short* Wl   = (unsigned short*)take((size_t)D * D * 2);
    (void)ws_size; (void)in_sizes; (void)n_in; (void)out_size;

    const long long nx = (long long)MR * D;          // 16.7M
    const dim3 ggrid(D / 128, MR / 128);             // (16, 64)
    const dim3 tgrid(D / 32, D / 32);                // weight transpose tiles

    // 1) x -> bf16 hi/lo
    split_bf16<<<(unsigned)(nx / 256), 256, 0, stream>>>(x, xh, xl, nx);

    // 2) q = x @ Wq
    split_transpose_bf16<<<tgrid, 256, 0, stream>>>(Wq, Wh, Wl, D, D);
    gemm_bf16x2<<<ggrid, 256, 0, stream>>>(xh, xl, Wh, Wl, qbuf, nullptr, MR, D, D);

    // 3) k = x @ Wk
    split_transpose_bf16<<<tgrid, 256, 0, stream>>>(Wk, Wh, Wl, D, D);
    gemm_bf16x2<<<ggrid, 256, 0, stream>>>(xh, xl, Wh, Wl, kbuf, nullptr, MR, D, D);

    // 4) feature-attention weights (softmax(Q^T K / sqrt(HO)))
    scores_softmax<<<dim3(HO, H, Bb), 128, 0, stream>>>(qbuf, kbuf, attw, Bb, T, D, H, HO);

    // 5) v = x @ Wv   (reuse k buffer)
    split_transpose_bf16<<<tgrid, 256, 0, stream>>>(Wv, Wh, Wl, D, D);
    gemm_bf16x2<<<ggrid, 256, 0, stream>>>(xh, xl, Wh, Wl, kbuf, nullptr, MR, D, D);

    // 6) ctx = weights x (v + alibi)   (reuse q buffer)
    ctx_kernel<<<dim3(T / CTX_TCH, H, Bb), 128, 0, stream>>>(attw, kbuf, qbuf, Bb, T, D, H, HO);

    // 7) out = ctx @ Wp + bp   (ctx bf16 split reuses x's bf16 buffers)
    split_bf16<<<(unsigned)(nx / 256), 256, 0, stream>>>(qbuf, xh, xl, nx);
    split_transpose_bf16<<<tgrid, 256, 0, stream>>>(Wp, Wh, Wl, D, D);
    gemm_bf16x2<<<ggrid, 256, 0, stream>>>(xh, xl, Wh, Wl, out, bp, MR, D, D);
}
